// GCRN_16183436772068
// MI455X (gfx1250) — matrix operand order repaired and checked by reading, not compiled, not run on hardware
//
#include <hip/hip_runtime.h>

// ---- problem constants (match reference) ----
#define B_   8
#define N_   1024
#define F_   128
#define G_   64
#define M_   4
#define E_   3
#define MEG_ 768          // M*E*G
#define RROWS_ 8192       // B*N
#define NEGV (-9.0e15f)

typedef __attribute__((ext_vector_type(16))) _Float16 v16h;
typedef __attribute__((ext_vector_type(8)))  _Float16 half8;
typedef __attribute__((ext_vector_type(8)))  float    v8f;

union HF16 { v16h v; half8 h[2]; };

__device__ __forceinline__ v8f wmma_step(v16h a, v16h b, v8f c) {
  // D = A(16x32 f16) x B(32x16 f16) + C(16x16 f32)
  return __builtin_amdgcn_wmma_f32_16x16x32_f16(
      /*neg_a=*/false, a, /*neg_b=*/false, b,
      /*c_mod=*/(short)0, c, /*reuse_a=*/false, /*reuse_b=*/false);
}

// A-frag from row-major f16 (rows = tile rows, 32 K-columns starting at Arow):
// lane l: m = l&15 ; elems 0..7 -> K = (l>>4)*8 + 0..7 ; elems 8..15 -> K = 16 + (l>>4)*8 + 0..7
__device__ __forceinline__ v16h load_a_frag(const _Float16* Arow_lane, int khalf) {
  HF16 a;
  a.h[0] = *(const half8*)(Arow_lane + khalf * 8);
  a.h[1] = *(const half8*)(Arow_lane + 16 + khalf * 8);
  return a.v;
}
// B-frag from K-contiguous (pre-transposed) f16: BTrow_lane points at [n=l&15] row of BT,
// lane l: elems 0..7 -> K = (l>>4)*8 + 0..7 ; elems 8..15 -> K = 16 + (l>>4)*8 + 0..7
__device__ __forceinline__ v16h load_b_frag(const _Float16* BTrow_lane, int khalf) {
  HF16 b;
  b.h[0] = *(const half8*)(BTrow_lane + khalf * 8);
  b.h[1] = *(const half8*)(BTrow_lane + 16 + khalf * 8);
  return b.v;
}

// ------------------- prep kernels -------------------
__global__ void k_cvt_f16(const float* __restrict__ in, _Float16* __restrict__ out, int n) {
  int i = blockIdx.x * 256 + threadIdx.x;
  if (i < n) out[i] = (_Float16)in[i];
}

// out[c*R + r] = in[r*C + c] per batch (blockIdx.y)
__global__ void k_tconv_f16(const float* __restrict__ in, _Float16* __restrict__ out,
                            int R, int C) {
  size_t base = (size_t)blockIdx.y * R * C;
  int i = blockIdx.x * 256 + threadIdx.x;
  if (i >= R * C) return;
  int r = i / C, c = i % C;
  out[base + (size_t)c * R + r] = (_Float16)in[base + i];
}

// ------------------- K1: Wh = X @ Wm, emit WhT (f16) + s1/s2 -------------------
// one wave per (b,e,m, 16-row tile); wave loops over 4 g-tiles.
__global__ void k1_wh(const _Float16* __restrict__ Xh, const _Float16* __restrict__ WsT,
                      const float* __restrict__ a1, const float* __restrict__ a2,
                      _Float16* __restrict__ WhT, float* __restrict__ s1,
                      float* __restrict__ s2) {
  int wave = (blockIdx.x * blockDim.x + threadIdx.x) >> 5;
  int lane = threadIdx.x & 31;
  int rt   = wave & 63;          // 64 row tiles of 16
  int rest = wave >> 6;          // ((b*E)+e)*M + hm
  int hm = rest % M_; int be = rest / M_;
  int e  = be % E_;   int b  = be / E_;
  int nl = lane & 15, khalf = lane >> 4;

  const _Float16* Arow = Xh + ((size_t)b * N_ + rt * 16 + nl) * F_;
  size_t whbase = ((size_t)(b * E_ + e) * M_ + hm) * (size_t)(G_ * N_);
  size_t sbase  = ((size_t)(b * E_ + e) * M_ + hm) * (size_t)N_;

  float s1acc[8], s2acc[8];
  for (int r = 0; r < 8; ++r) { s1acc[r] = 0.f; s2acc[r] = 0.f; }

  for (int gt = 0; gt < 4; ++gt) {
    const _Float16* Brow = WsT + ((size_t)(e * M_ + hm) * G_ + gt * 16 + nl) * F_;
    v8f acc = {};
    for (int k = 0; k < F_; k += 32)
      acc = wmma_step(load_a_frag(Arow + k, khalf), load_b_frag(Brow + k, khalf), acc);

    // store transposed: WhT[g][j], rows j contiguous per lane -> one b128 store
    half8 hh;
    for (int r = 0; r < 8; ++r) hh[r] = (_Float16)acc[r];
    *(half8*)(WhT + whbase + (size_t)(gt * 16 + nl) * N_ + rt * 16 + khalf * 8) = hh;

    float a1v = a1[(size_t)e * (M_ * G_) + hm * G_ + gt * 16 + nl];
    float a2v = a2[(size_t)e * (M_ * G_) + hm * G_ + gt * 16 + nl];
    for (int r = 0; r < 8; ++r) { s1acc[r] += acc[r] * a1v; s2acc[r] += acc[r] * a2v; }
  }
  // reduce over the 16 lanes that share a row (g dimension)
  for (int r = 0; r < 8; ++r) {
    float v1 = s1acc[r], v2 = s2acc[r];
    for (int o = 8; o; o >>= 1) { v1 += __shfl_xor(v1, o, 16); v2 += __shfl_xor(v2, o, 16); }
    if (nl == 0) {
      size_t idx = sbase + rt * 16 + khalf * 8 + r;
      s1[idx] = v1; s2[idx] = v2;
    }
  }
}

// ------------------- K2: masked softmax + Hm = elu(att @ Wh) -------------------
// block per (b,e,m, 16-row tile); 256 threads = 8 waves.
__global__ void k2_att(const int* __restrict__ A, const float* __restrict__ s1,
                       const float* __restrict__ s2, const _Float16* __restrict__ WhT,
                       _Float16* __restrict__ Hh) {
  __shared__ float    s2sh[N_];
  __shared__ float    s1sh[16], rmax[16], rsum[16];
  __shared__ _Float16 Psh[16 * N_];          // unnormalized probs (f16)
  __shared__ float    red[4][32][8];

  int blk = blockIdx.x;
  int it = blk & 63; int rest = blk >> 6;
  int hm = rest % M_; int be = rest / M_;
  int e  = be % E_;   int b  = be / E_;
  int t = threadIdx.x;
  size_t sbase  = ((size_t)(b * E_ + e) * M_ + hm) * (size_t)N_;
  size_t whbase = ((size_t)(b * E_ + e) * M_ + hm) * (size_t)(G_ * N_);

  for (int j = t; j < N_; j += 256) s2sh[j] = s2[sbase + j];
  if (t < 16) s1sh[t] = s1[sbase + it * 16 + t];
  __syncthreads();

  // ---- phase A: row max / sum-exp, write P tile to LDS ----
  int row = t >> 4, jl = t & 15;
  const int* Ar = A + (((size_t)b * E_ + e) * N_ + it * 16 + row) * N_;
  float s1v = s1sh[row];
  float lmax = -3.0e38f;
  for (int j = jl; j < N_; j += 16) {
    float sc = s1v + s2sh[j];
    sc = sc > 0.f ? sc : 0.01f * sc;          // leaky_relu(0.01)
    sc = (Ar[j] > 0) ? sc : NEGV;             // mask
    lmax = fmaxf(lmax, sc);
  }
  for (int o = 8; o; o >>= 1) lmax = fmaxf(lmax, __shfl_xor(lmax, o, 16));
  if (jl == 0) rmax[row] = lmax;
  __syncthreads();
  float mx = rmax[row];
  float lsum = 0.f;
  for (int j = jl; j < N_; j += 16) {
    float sc = s1v + s2sh[j];
    sc = sc > 0.f ? sc : 0.01f * sc;
    sc = (Ar[j] > 0) ? sc : NEGV;
    float ev = expf(sc - mx);
    Psh[row * N_ + j] = (_Float16)ev;
    lsum += ev;
  }
  for (int o = 8; o; o >>= 1) lsum += __shfl_xor(lsum, o, 16);
  if (jl == 0) rsum[row] = lsum;
  __syncthreads();

  // ---- phase B: (16x1024 P) @ (1024x64 Wh) via WMMA, split over 8 waves ----
  int w = t >> 5, lane = t & 31;
  int gt = w & 3, jseg = w >> 2;              // 2 waves per g-tile, 512 K each
  int nl = lane & 15, khalf = lane >> 4;
  const _Float16* Prow = Psh + (size_t)nl * N_ + jseg * 512;
  const _Float16* Brow = WhT + whbase + (size_t)(gt * 16 + nl) * N_ + jseg * 512;
  v8f acc = {};
  for (int k = 0; k < 512; k += 32)
    acc = wmma_step(load_a_frag(Prow + k, khalf), load_b_frag(Brow + k, khalf), acc);

  if (w >= 4) { for (int r = 0; r < 8; ++r) red[gt][lane][r] = acc[r]; }
  __syncthreads();
  if (w < 4) {
    for (int r = 0; r < 8; ++r) {
      int irow = khalf * 8 + r;
      float v = (acc[r] + red[gt][lane][r]) * (1.0f / rsum[irow]);   // normalize
      v = v > 0.f ? v : (expf(v) - 1.0f);                            // elu
      // H column layout: hm*(E*G) + e*G + g
      Hh[((size_t)b * N_ + it * 16 + irow) * MEG_ + hm * (E_ * G_) + e * G_ + gt * 16 + nl] =
          (_Float16)v;
    }
  }
}

// ------------------- generic WMMA GEMM: C(f32) = A(f16,row-major) @ BT(f16,[N][K]) ----
__global__ void k_gemm(const _Float16* __restrict__ Af, const _Float16* __restrict__ BT,
                       float* __restrict__ C, int Mr, int Nc, int K) {
  int wave = (blockIdx.x * blockDim.x + threadIdx.x) >> 5;
  int lane = threadIdx.x & 31;
  int ntn = Nc >> 4;
  if (wave >= (Mr >> 4) * ntn) return;
  int tm = wave / ntn, tn = wave % ntn;
  int nl = lane & 15, khalf = lane >> 4;
  const _Float16* Arow = Af + ((size_t)tm * 16 + nl) * K;
  const _Float16* Brow = BT + ((size_t)tn * 16 + nl) * K;
  v8f acc = {};
  for (int k = 0; k < K; k += 32) {
    __builtin_prefetch(Arow + k + 64, 0, 0);
    __builtin_prefetch(Brow + k + 64, 0, 0);
    acc = wmma_step(load_a_frag(Arow + k, khalf), load_b_frag(Brow + k, khalf), acc);
  }
  float* Cp = C + ((size_t)tm * 16 + khalf * 8) * Nc + tn * 16 + nl;
  for (int r = 0; r < 8; ++r) Cp[(size_t)r * Nc] = acc[r];
}

// ------------------- batch-norm helpers -------------------
__global__ void k_colstats(const float* __restrict__ X, int R, int Cc,
                           float* __restrict__ mean, float* __restrict__ var) {
  int c = blockIdx.x, t = threadIdx.x;
  float s = 0.f, q = 0.f;
  for (int r = t; r < R; r += 256) { float v = X[(size_t)r * Cc + c]; s += v; q += v * v; }
  __shared__ float ss[256], qq[256];
  ss[t] = s; qq[t] = q; __syncthreads();
  for (int o = 128; o; o >>= 1) { if (t < o) { ss[t] += ss[t + o]; qq[t] += qq[t + o]; } __syncthreads(); }
  if (t == 0) { float m = ss[0] / R; mean[c] = m; var[c] = qq[0] / R - m * m; }
}

__global__ void k_bn_apply(const float* __restrict__ X, const float* __restrict__ mean,
                           const float* __restrict__ var, const float* __restrict__ g,
                           const float* __restrict__ bta, int R, int Cc,
                           float* __restrict__ outf, _Float16* __restrict__ outh, int do_elu) {
  size_t i = (size_t)blockIdx.x * 256 + threadIdx.x;
  if (i >= (size_t)R * Cc) return;
  int c = (int)(i % Cc);
  float v = g[c] * (X[i] - mean[c]) * rsqrtf(var[c] + 1e-5f) + bta[c];
  if (do_elu) v = v > 0.f ? v : (expf(v) - 1.0f);
  if (outf) outf[i] = v;
  if (outh) outh[i] = (_Float16)v;
}

__global__ void k_mix(float* __restrict__ t, const float* __restrict__ x, int n) {
  int i = blockIdx.x * 256 + threadIdx.x;
  if (i < n) t[i] = 0.5f * t[i] + 0.5f * x[i];     // (1-ALPHA)*G + ALPHA*X, ALPHA=0.5
}
__global__ void k_addin(float* __restrict__ t, const float* __restrict__ h, int n) {
  int i = blockIdx.x * 256 + threadIdx.x;
  if (i < n) t[i] += h[i];
}

// ------------------- edge_cat_tensor: transpose(A,(0,2,3,1)) as f32 -------------------
__global__ void k_edge(const int* __restrict__ A, float* __restrict__ out) {
  size_t i = (size_t)blockIdx.x * 256 + threadIdx.x;       // over B*N*N
  if (i >= (size_t)B_ * N_ * N_) return;
  size_t j = i % N_; size_t bi = i / N_;
  size_t row = bi % N_; size_t b = bi / N_;
  float* o = out + i * E_;
  for (int e = 0; e < E_; ++e)
    o[e] = (float)A[(((size_t)b * E_ + e) * N_ + row) * N_ + j];
}

// ------------------- host orchestration -------------------
extern "C" void kernel_launch(void* const* d_in, const int* in_sizes, int n_in,
                              void* d_out, int out_size, void* d_ws, size_t ws_size,
                              hipStream_t stream) {
  const int*   A    = (const int*)  d_in[0];
  const float* X    = (const float*)d_in[1];
  const float* Ws   = (const float*)d_in[3];
  const float* a1   = (const float*)d_in[4];
  const float* a2   = (const float*)d_in[5];
  const float* W1   = (const float*)d_in[6];
  const float* bn1g = (const float*)d_in[7];
  const float* bn1b = (const float*)d_in[8];
  const float* el0  = (const float*)d_in[9];
  const float* bg0  = (const float*)d_in[10];
  const float* bb0  = (const float*)d_in[11];
  const float* el1  = (const float*)d_in[12];
  const float* bg1  = (const float*)d_in[13];
  const float* bb1  = (const float*)d_in[14];
  const float* el2  = (const float*)d_in[15];
  const float* bn2g = (const float*)d_in[16];
  const float* bn2b = (const float*)d_in[17];
  (void)in_sizes; (void)n_in; (void)out_size; (void)ws_size;

  char* wsb = (char*)d_ws; size_t off = 0;
  auto alloc = [&](size_t bytes) -> void* {
    void* p = wsb + off; off += (bytes + 255) & ~(size_t)255; return p;
  };
  _Float16* Xh   = (_Float16*)alloc((size_t)RROWS_ * F_ * 2);
  _Float16* WsT  = (_Float16*)alloc((size_t)E_ * M_ * G_ * F_ * 2);
  _Float16* W1T  = (_Float16*)alloc((size_t)F_ * MEG_ * 2);
  _Float16* e0T  = (_Float16*)alloc((size_t)256 * F_ * 2);
  _Float16* e1T  = (_Float16*)alloc((size_t)256 * 256 * 2);
  _Float16* e2T  = (_Float16*)alloc((size_t)F_ * 256 * 2);
  _Float16* WhT  = (_Float16*)alloc((size_t)B_ * E_ * M_ * G_ * N_ * 2);
  float*    s1   = (float*)   alloc((size_t)B_ * E_ * M_ * N_ * 4);
  float*    s2   = (float*)   alloc((size_t)B_ * E_ * M_ * N_ * 4);
  _Float16* Hh   = (_Float16*)alloc((size_t)RROWS_ * MEG_ * 2);
  float*    T1   = (float*)   alloc((size_t)RROWS_ * F_ * 4);
  float*    Hb   = (float*)   alloc((size_t)RROWS_ * F_ * 4);
  _Float16* Hbh  = (_Float16*)alloc((size_t)RROWS_ * F_ * 2);
  float*    G2   = (float*)   alloc((size_t)RROWS_ * 256 * 4);
  _Float16* Z0h  = (_Float16*)alloc((size_t)RROWS_ * 256 * 2);
  float*    G3   = (float*)   alloc((size_t)RROWS_ * 256 * 4);
  _Float16* Z1h  = (_Float16*)alloc((size_t)RROWS_ * 256 * 2);
  float*    Z2   = (float*)   alloc((size_t)RROWS_ * F_ * 4);
  float*    mn   = (float*)   alloc(256 * 4);
  float*    vr   = (float*)   alloc(256 * 4);

  float* Zout    = (float*)d_out;
  float* EdgeOut = Zout + (size_t)B_ * N_ * F_;

  auto gemm = [&](const _Float16* Af, const _Float16* BT, float* C, int Mr, int Nc, int K) {
    int waves = (Mr / 16) * (Nc / 16);
    k_gemm<<<dim3((waves + 7) / 8), dim3(256), 0, stream>>>(Af, BT, C, Mr, Nc, K);
  };

  // prep: f16 conversions / weight transposes
  k_cvt_f16<<<dim3((RROWS_ * F_ + 255) / 256), dim3(256), 0, stream>>>(X, Xh, RROWS_ * F_);
  k_tconv_f16<<<dim3((F_ * G_ + 255) / 256, E_ * M_), dim3(256), 0, stream>>>(Ws, WsT, F_, G_);
  k_tconv_f16<<<dim3((MEG_ * F_ + 255) / 256, 1), dim3(256), 0, stream>>>(W1, W1T, MEG_, F_);
  k_tconv_f16<<<dim3((F_ * 256 + 255) / 256, 1), dim3(256), 0, stream>>>(el0, e0T, F_, 256);
  k_tconv_f16<<<dim3((256 * 256 + 255) / 256, 1), dim3(256), 0, stream>>>(el1, e1T, 256, 256);
  k_tconv_f16<<<dim3((256 * F_ + 255) / 256, 1), dim3(256), 0, stream>>>(el2, e2T, 256, F_);

  // K1: Wh + s1/s2   (6144 waves -> 768 blocks of 8 waves)
  k1_wh<<<dim3(768), dim3(256), 0, stream>>>(Xh, WsT, a1, a2, WhT, s1, s2);
  // K2: masked softmax + att@Wh + elu -> H (f16)
  k2_att<<<dim3(B_ * E_ * M_ * 64), dim3(256), 0, stream>>>(A, s1, s2, WhT, Hh);

  // H @ W1 -> T1 ; mix with X ; bn1 -> Hb (f32+f16)
  gemm(Hh, W1T, T1, RROWS_, F_, MEG_);
  k_mix<<<dim3((RROWS_ * F_ + 255) / 256), dim3(256), 0, stream>>>(T1, X, RROWS_ * F_);
  k_colstats<<<dim3(F_), dim3(256), 0, stream>>>(T1, RROWS_, F_, mn, vr);
  k_bn_apply<<<dim3((RROWS_ * F_ + 255) / 256), dim3(256), 0, stream>>>(
      T1, mn, vr, bn1g, bn1b, RROWS_, F_, Hb, Hbh, 0);

  // encoder: elu(bn(Hb@e0)) -> Z0 ; elu(bn(Z0@e1)) -> Z1 ; Z1@e2 -> Z2
  gemm(Hbh, e0T, G2, RROWS_, 256, F_);
  k_colstats<<<dim3(256), dim3(256), 0, stream>>>(G2, RROWS_, 256, mn, vr);
  k_bn_apply<<<dim3((RROWS_ * 256 + 255) / 256), dim3(256), 0, stream>>>(
      G2, mn, vr, bg0, bb0, RROWS_, 256, (float*)nullptr, Z0h, 1);

  gemm(Z0h, e1T, G3, RROWS_, 256, 256);
  k_colstats<<<dim3(256), dim3(256), 0, stream>>>(G3, RROWS_, 256, mn, vr);
  k_bn_apply<<<dim3((RROWS_ * 256 + 255) / 256), dim3(256), 0, stream>>>(
      G3, mn, vr, bg1, bb1, RROWS_, 256, (float*)nullptr, Z1h, 1);

  gemm(Z1h, e2T, Z2, RROWS_, F_, 256);

  // Z = bn(Hb + Z2) -> d_out
  k_addin<<<dim3((RROWS_ * F_ + 255) / 256), dim3(256), 0, stream>>>(Z2, Hb, RROWS_ * F_);
  k_colstats<<<dim3(F_), dim3(256), 0, stream>>>(Z2, RROWS_, F_, mn, vr);
  k_bn_apply<<<dim3((RROWS_ * F_ + 255) / 256), dim3(256), 0, stream>>>(
      Z2, mn, vr, bn2g, bn2b, RROWS_, F_, Zout, (_Float16*)nullptr, 0);

  // edge_cat_tensor
  k_edge<<<dim3((B_ * N_ * N_ + 255) / 256), dim3(256), 0, stream>>>(A, EdgeOut);
}
